// MultiHeadAttention_89893665506154
// MI455X (gfx1250) — compile-verified, matrix operand order checked by reading, not run on hardware
//
#include <hip/hip_runtime.h>
#include <hip/hip_bf16.h>
#include <stdint.h>

// MHA forward, MI455X (gfx1250), bf16 WMMA everywhere, double-buffered
// async global->LDS staging (ASYNCcnt pipelining).
// Workspace (~66 MB): Xbuf (16 MB, reused: q/k/v bf16, then attn out) |
// Wbuf (2 MB, reused: Wq/Wk/Wv/Wo bf16) | Qb | Kb | Vb (head-major bf16).

#define DM   1024
#define HH   16
#define DK   64
#define BB   4
#define SS   2048
#define MTOT (BB * SS)

typedef __attribute__((ext_vector_type(16))) __bf16 v16bf;
typedef __attribute__((ext_vector_type(8)))  float  v8f;

// ---- fp32 -> bf16 packing (hardware cvt_pk when available) ----------------
#if __has_builtin(__builtin_amdgcn_cvt_pk_bf16_f32)
typedef __attribute__((ext_vector_type(2))) __bf16 v2bf;
__device__ __forceinline__ unsigned pk2(float a, float b) {
  union { v2bf v; unsigned u; } t;
  t.v = __builtin_amdgcn_cvt_pk_bf16_f32(a, b);
  return t.u;
}
#else
__device__ __forceinline__ unsigned short f2bf_sw(float f) {
  unsigned u = __float_as_uint(f);
  u += 0x7FFFu + ((u >> 16) & 1u);
  return (unsigned short)(u >> 16);
}
__device__ __forceinline__ unsigned pk2(float a, float b) {
  return (unsigned)f2bf_sw(a) | ((unsigned)f2bf_sw(b) << 16);
}
#endif
__device__ __forceinline__ unsigned short f2bf(float f) {
  return (unsigned short)pk2(f, 0.f);
}

// ---- gfx1250 async global->LDS staging ------------------------------------
__device__ __forceinline__ unsigned lds_addr32(const void* p) {
  // generic pointer to LDS: addr[31:0] is the LDS byte offset
  return (unsigned)(uintptr_t)p;
}
__device__ __forceinline__ void async_cp16(unsigned lds, const void* g) {
  asm volatile("global_load_async_to_lds_b128 %0, %1, off"
               :: "v"(lds), "v"((unsigned long long)(uintptr_t)g)
               : "memory");
}
template <int N>
__device__ __forceinline__ void async_wait_le() {
#if __has_builtin(__builtin_amdgcn_s_wait_asynccnt)
  __builtin_amdgcn_s_wait_asynccnt(N);
#else
  asm volatile("s_wait_asynccnt %0" :: "i"(N) : "memory");
#endif
}

// Load one 16x32 bf16 WMMA operand fragment from LDS (CDNA5 ISA layout):
// lanes 0-15: K=0..7 (VGPR0-3) + K=16..23 (VGPR4-7); lanes 16-31: K=8..15 +
// K=24..31. Row = lane&15. Used for A (row=M) and B (row=N, K-contiguous).
__device__ __forceinline__ v16bf lds_frag(const unsigned short* base, int row,
                                          int stride, int kbase) {
  const int lane = threadIdx.x & 31;
  const int r = row + (lane & 15);
  const int ko = kbase + ((lane >> 4) << 3);
  const unsigned short* p = base + r * stride + ko;   // 16B-aligned by design
  union { uint4 u[2]; v16bf v; } f;
  f.u[0] = *(const uint4*)(p);
  f.u[1] = *(const uint4*)(p + 16);
  return f.v;
}

__device__ __forceinline__ v8f wmma_bf16(v16bf a, v16bf b, v8f c) {
  return __builtin_amdgcn_wmma_f32_16x16x32_bf16(false, a, false, b,
                                                 (short)0, c, false, false);
}

// ---------------------------------------------------------------------------
// One-shot fp32 -> bf16 conversion (4 elems / thread).
// ---------------------------------------------------------------------------
__global__ __launch_bounds__(256) void f32_to_bf16_kernel(
    const float* __restrict__ x, unsigned short* __restrict__ y, int n4) {
  int i = blockIdx.x * 256 + threadIdx.x;
  if (i < n4) {
    float4 v = *(const float4*)(x + (size_t)i * 4);
    uint2 t; t.x = pk2(v.x, v.y); t.y = pk2(v.z, v.w);
    *(uint2*)(y + (size_t)i * 4) = t;
  }
}

// ---------------------------------------------------------------------------
// Pure-bf16 GEMM: Y = X[M,1024] @ W[1024,1024]^T + bias.
// Block tile 256x128, BK=32, 256 threads = 8 waves (4M x 2N), wave 64x64.
// Double-buffered async staging: issue tile t+1 (6 copies/thread), then
// s_wait_asynccnt 6 proves tile t landed (async loads complete in order).
// OUTF32==0: bf16 out, head-major [B,H,S,DK]. OUTF32==1: fp32 row-major.
// ---------------------------------------------------------------------------
__device__ __forceinline__ void gemm_stage(
    const unsigned short* __restrict__ X, const unsigned short* __restrict__ W,
    int rowBase, int colBase, int kt, unsigned asB, unsigned bsB, int tid) {
#pragma unroll
  for (int i = 0; i < 4; ++i) {          // A: 256 rows x 64 B, 4 chunks/thread
    int slot = tid + i * 256;
    int r = slot >> 2, c = slot & 3;
    async_cp16(asB + (unsigned)(r * 40 + c * 8) * 2,
               X + (size_t)(rowBase + r) * DM + kt + c * 8);
  }
#pragma unroll
  for (int i = 0; i < 2; ++i) {          // B: 128 rows x 64 B, 2 chunks/thread
    int slot = tid + i * 256;
    int r = slot >> 2, c = slot & 3;
    async_cp16(bsB + (unsigned)(r * 40 + c * 8) * 2,
               W + (size_t)(colBase + r) * DM + kt + c * 8);
  }
}

template <int OUTF32>
__global__ __launch_bounds__(256) void gemm_bf16_kernel(
    const unsigned short* __restrict__ X, const unsigned short* __restrict__ W,
    const float* __restrict__ bias, void* __restrict__ Yv) {
  __shared__ unsigned short As[2 * 256 * 40];   // stride 40 bf16 = 80 B
  __shared__ unsigned short Bs[2 * 128 * 40];

  const int tid  = threadIdx.x;
  const int lane = tid & 31;
  const int wave = tid >> 5;
  const int wm = (wave & 3) * 64;
  const int wn = (wave >> 2) * 64;
  const int rowBase = blockIdx.y * 256;
  const int colBase = blockIdx.x * 128;
  const unsigned asB = lds_addr32(As);
  const unsigned bsB = lds_addr32(Bs);

  v8f acc[4][4];
#pragma unroll
  for (int i = 0; i < 4; ++i)
#pragma unroll
    for (int j = 0; j < 4; ++j) acc[i][j] = {};

  const int NIT = DM / 32;   // 32
  gemm_stage(X, W, rowBase, colBase, 0, asB, bsB, tid);   // prologue -> buf0

  for (int it = 0; it < NIT; ++it) {
    const int cur = it & 1;
    const unsigned short* Ac = As + cur * (256 * 40);
    const unsigned short* Bc = Bs + cur * (128 * 40);
    if (it + 1 < NIT) {
      gemm_stage(X, W, rowBase, colBase, (it + 1) * 32,
                 asB + (cur ^ 1) * (256 * 40 * 2),
                 bsB + (cur ^ 1) * (128 * 40 * 2), tid);
      async_wait_le<6>();    // tile `it` complete; tile `it+1` in flight
    } else {
      async_wait_le<0>();
    }
    __syncthreads();

    v16bf af[4], bfr[4];
#pragma unroll
    for (int i = 0; i < 4; ++i) af[i] = lds_frag(Ac, wm + i * 16, 40, 0);
#pragma unroll
    for (int j = 0; j < 4; ++j) bfr[j] = lds_frag(Bc, wn + j * 16, 40, 0);
#pragma unroll
    for (int i = 0; i < 4; ++i)
#pragma unroll
      for (int j = 0; j < 4; ++j)
        acc[i][j] = wmma_bf16(af[i], bfr[j], acc[i][j]);
    __syncthreads();
  }

  // C layout: VGPR e -> row e (lanes 0-15) / e+8 (lanes 16-31), col = lane&15.
  const int halfoff = (lane >> 4) << 3;
#pragma unroll
  for (int i = 0; i < 4; ++i)
#pragma unroll
    for (int j = 0; j < 4; ++j) {
      int gc = colBase + wn + j * 16 + (lane & 15);
      float bv = bias[gc];
      if (OUTF32) {
        float* Y = (float*)Yv;
#pragma unroll
        for (int e = 0; e < 8; ++e) {
          int gr = rowBase + wm + i * 16 + e + halfoff;
          Y[(size_t)gr * DM + gc] = acc[i][j][e] + bv;
        }
      } else {
        unsigned short* Y = (unsigned short*)Yv;
        int h = gc >> 6, d = gc & 63;
#pragma unroll
        for (int e = 0; e < 8; ++e) {
          int gr = rowBase + wm + i * 16 + e + halfoff;
          int b = gr >> 11, s = gr & (SS - 1);
          Y[(((size_t)(b * HH + h) * SS) + s) * DK + d] =
              f2bf(acc[i][j][e] + bv);
        }
      }
    }
}

// ---------------------------------------------------------------------------
// Flash attention: one block = 128 q-rows of one (b,h); 8 waves, each wave
// owns 16 rows end-to-end (intra-wave online softmax). Double-buffered K
// (async to LDS) and V (global->VGPR before compute, VGPR->LDS-transposed
// after compute, published by the closing barrier). Writes O (/l) bf16
// row-major [B,S,1024].
// ---------------------------------------------------------------------------
__device__ __forceinline__ void attn_stage_k(
    const unsigned short* __restrict__ Kb, size_t headBase, int k0,
    unsigned ksB, int tid) {
#pragma unroll
  for (int i = 0; i < 4; ++i) {          // 128 rows x 128 B = 1024 chunks
    int slot = tid + i * 256;
    int r = slot >> 3, c = slot & 7;
    async_cp16(ksB + (unsigned)(r * 72 + c * 8) * 2,
               Kb + headBase + (size_t)(k0 + r) * DK + c * 8);
  }
}

__global__ __launch_bounds__(256) void flash_attn_kernel(
    const unsigned short* __restrict__ Qb, const unsigned short* __restrict__ Kb,
    const unsigned short* __restrict__ Vb, unsigned short* __restrict__ Ob) {
  __shared__ unsigned short Qs[128 * 72];         // [qrow][d], stride 72
  __shared__ unsigned short Ks[2 * 128 * 72];     // [key][d], double buffer
  __shared__ unsigned short Vt[2 * 64 * 136];     // [d][key], double buffer
  __shared__ unsigned short Ps[8 * 16 * 136];     // per-wave P strip

  const int tid  = threadIdx.x;
  const int lane = tid & 31;
  const int wave = tid >> 5;
  const int bh = blockIdx.y;                      // b*H + h
  const int b = bh >> 4, h = bh & 15;
  const int q0 = blockIdx.x * 128;
  const size_t headBase = (size_t)bh * SS * DK;
  const unsigned qsB = lds_addr32(Qs);
  const unsigned ksB = lds_addr32(Ks);

  // Q tile once (async).
#pragma unroll
  for (int i = 0; i < 4; ++i) {
    int slot = tid + i * 256;
    int r = slot >> 3, c = slot & 7;
    async_cp16(qsB + (unsigned)(r * 72 + c * 8) * 2,
               Qb + headBase + (size_t)(q0 + r) * DK + c * 8);
  }
  // Prologue: K0 async; V0 loaded and transposed immediately.
  attn_stage_k(Kb, headBase, 0, ksB, tid);
  {
#pragma unroll
    for (int i = 0; i < 8; ++i) {
      int slot = tid + i * 256;                   // 2048 4-elem slots
      int key = slot >> 4;
      int d = (slot & 15) * 4;
      uint2 x = *(const uint2*)(Vb + headBase + (size_t)key * DK + d);
      Vt[(d + 0) * 136 + key] = (unsigned short)(x.x);
      Vt[(d + 1) * 136 + key] = (unsigned short)(x.x >> 16);
      Vt[(d + 2) * 136 + key] = (unsigned short)(x.y);
      Vt[(d + 3) * 136 + key] = (unsigned short)(x.y >> 16);
    }
  }

  float mrow[8], lrow[8];
  v8f oacc[4];
#pragma unroll
  for (int e = 0; e < 8; ++e) { mrow[e] = -1e30f; lrow[e] = 0.f; }
#pragma unroll
  for (int t = 0; t < 4; ++t) oacc[t] = {};

  v16bf aQ0, aQ1;
  const float cl2 = 0.125f * 1.44269504088896f;   // (1/sqrt(64)) * log2(e)
  const int NT = SS / 128;                        // 16

  for (int t = 0; t < NT; ++t) {
    const int cur = t & 1, nxt = cur ^ 1;
    const unsigned short* KsC = Ks + cur * (128 * 72);
    const unsigned short* VtC = Vt + cur * (64 * 136);
    const bool hasNext = (t + 1 < NT);

    uint2 vreg[8];
    if (hasNext) {
      // K(t+1) async into the other buffer; V(t+1) into VGPRs (latency hides
      // under this tile's WMMAs; LDS scatter happens after compute).
      attn_stage_k(Kb, headBase, (t + 1) * 128, ksB + nxt * (128 * 72 * 2), tid);
#pragma unroll
      for (int i = 0; i < 8; ++i) {
        int slot = tid + i * 256;
        int key = slot >> 4;
        int d = (slot & 15) * 4;
        vreg[i] = *(const uint2*)(Vb + headBase +
                                  (size_t)((t + 1) * 128 + key) * DK + d);
      }
      async_wait_le<4>();   // K(t) (and Q on t==0) complete, K(t+1) in flight
    } else {
      async_wait_le<0>();
    }
    __syncthreads();

    if (t == 0) {
      aQ0 = lds_frag(Qs, wave * 16, 72, 0);
      aQ1 = lds_frag(Qs, wave * 16, 72, 32);
    }

    // S = Q K^T for this wave's 16 rows x 128 keys (8 N-tiles, K=64).
    v8f st[8];
#pragma unroll
    for (int tt = 0; tt < 8; ++tt) {
      v8f z = {};
      z = wmma_bf16(aQ0, lds_frag(KsC, tt * 16, 72, 0), z);
      z = wmma_bf16(aQ1, lds_frag(KsC, tt * 16, 72, 32), z);
      st[tt] = z;
    }

    // Online softmax: row e lives in VGPR e across a 16-lane half.
#pragma unroll
    for (int e = 0; e < 8; ++e) {
      float mx = -1e30f;
#pragma unroll
      for (int tt = 0; tt < 8; ++tt) mx = fmaxf(mx, st[tt][e]);
      mx = fmaxf(mx, __shfl_xor(mx, 1));
      mx = fmaxf(mx, __shfl_xor(mx, 2));
      mx = fmaxf(mx, __shfl_xor(mx, 4));
      mx = fmaxf(mx, __shfl_xor(mx, 8));
      float mnew = fmaxf(mrow[e], mx * cl2);
      float alpha = exp2f(mrow[e] - mnew);
      mrow[e] = mnew;
      float rs = 0.f;
#pragma unroll
      for (int tt = 0; tt < 8; ++tt) {
        float p = exp2f(st[tt][e] * cl2 - mnew);
        st[tt][e] = p;
        rs += p;
      }
      rs += __shfl_xor(rs, 1);
      rs += __shfl_xor(rs, 2);
      rs += __shfl_xor(rs, 4);
      rs += __shfl_xor(rs, 8);
      lrow[e] = lrow[e] * alpha + rs;
#pragma unroll
      for (int t2 = 0; t2 < 4; ++t2) oacc[t2][e] *= alpha;
    }

    // C-layout -> A-layout via wave-private LDS strip.
    unsigned short* Pw = &Ps[wave * 16 * 136];
    const int prow = (lane >> 4) << 3;
#pragma unroll
    for (int e = 0; e < 8; ++e)
#pragma unroll
      for (int tt = 0; tt < 8; ++tt)
        Pw[(e + prow) * 136 + tt * 16 + (lane & 15)] = f2bf(st[tt][e]);

    // O += P @ V  (K = 128 keys = 4 WMMA k-steps).
#pragma unroll
    for (int kk = 0; kk < 4; ++kk) {
      v16bf aP = lds_frag(Pw, 0, 136, kk * 32);
#pragma unroll
      for (int t2 = 0; t2 < 4; ++t2)
        oacc[t2] = wmma_bf16(aP, lds_frag(VtC, t2 * 16, 136, kk * 32), oacc[t2]);
    }

    // Scatter V(t+1) transpose into the free buffer; published by the
    // closing barrier (its implicit s_wait_dscnt 0).
    if (hasNext) {
      unsigned short* VtN = Vt + nxt * (64 * 136);
#pragma unroll
      for (int i = 0; i < 8; ++i) {
        int slot = tid + i * 256;
        int key = slot >> 4;
        int d = (slot & 15) * 4;
        VtN[(d + 0) * 136 + key] = (unsigned short)(vreg[i].x);
        VtN[(d + 1) * 136 + key] = (unsigned short)(vreg[i].x >> 16);
        VtN[(d + 2) * 136 + key] = (unsigned short)(vreg[i].y);
        VtN[(d + 3) * 136 + key] = (unsigned short)(vreg[i].y >> 16);
      }
    }
    __syncthreads();
  }

  // Epilogue: O / l, bf16, row-major [B,S,1024], column block h*64.
  const int halfoff = (lane >> 4) << 3;
#pragma unroll
  for (int e = 0; e < 8; ++e) {
    float inv = 1.0f / lrow[e];
    int sg = q0 + wave * 16 + e + halfoff;
#pragma unroll
    for (int t2 = 0; t2 < 4; ++t2) {
      int col = h * DK + t2 * 16 + (lane & 15);
      Ob[((size_t)(b * SS + sg)) * DM + col] = f2bf(oacc[t2][e] * inv);
    }
  }
}

// ---------------------------------------------------------------------------
extern "C" void kernel_launch(void* const* d_in, const int* in_sizes, int n_in,
                              void* d_out, int out_size, void* d_ws, size_t ws_size,
                              hipStream_t stream) {
  const float* q  = (const float*)d_in[0];
  const float* k  = (const float*)d_in[1];
  const float* v  = (const float*)d_in[2];
  const float* Wq = (const float*)d_in[3];
  const float* bq = (const float*)d_in[4];
  const float* Wk = (const float*)d_in[5];
  const float* bk = (const float*)d_in[6];
  const float* Wv = (const float*)d_in[7];
  const float* bv = (const float*)d_in[8];
  const float* Wo = (const float*)d_in[9];
  const float* bo = (const float*)d_in[10];
  float* out = (float*)d_out;

  // Workspace layout (all bf16): Xbuf 16 MB (reused 4x) | Wbuf 2 MB (reused
  // 4x) | Qb | Kb | Vb 16 MB each. Total 66 MB.
  unsigned short* Xbuf = (unsigned short*)d_ws;
  unsigned short* Wbuf = Xbuf + (size_t)MTOT * DM;
  unsigned short* Qb   = Wbuf + (size_t)DM * DM;
  unsigned short* Kb   = Qb + (size_t)MTOT * DM;
  unsigned short* Vb   = Kb + (size_t)MTOT * DM;

  const int nx4 = MTOT * DM / 4;      // 2,097,152
  const int nw4 = DM * DM / 4;        // 262,144
  dim3 gGemm(DM / 128, MTOT / 256);   // (8, 32)

  // Q projection
  f32_to_bf16_kernel<<<nx4 / 256, 256, 0, stream>>>(q, Xbuf, nx4);
  f32_to_bf16_kernel<<<nw4 / 256, 256, 0, stream>>>(Wq, Wbuf, nw4);
  gemm_bf16_kernel<0><<<gGemm, 256, 0, stream>>>(Xbuf, Wbuf, bq, Qb);
  // K projection
  f32_to_bf16_kernel<<<nx4 / 256, 256, 0, stream>>>(k, Xbuf, nx4);
  f32_to_bf16_kernel<<<nw4 / 256, 256, 0, stream>>>(Wk, Wbuf, nw4);
  gemm_bf16_kernel<0><<<gGemm, 256, 0, stream>>>(Xbuf, Wbuf, bk, Kb);
  // V projection
  f32_to_bf16_kernel<<<nx4 / 256, 256, 0, stream>>>(v, Xbuf, nx4);
  f32_to_bf16_kernel<<<nw4 / 256, 256, 0, stream>>>(Wv, Wbuf, nw4);
  gemm_bf16_kernel<0><<<gGemm, 256, 0, stream>>>(Xbuf, Wbuf, bv, Vb);

  // Attention (writes bf16 row-major into Xbuf, which is now free).
  dim3 gAttn(SS / 128, BB * HH);      // (16, 64)
  flash_attn_kernel<<<gAttn, 256, 0, stream>>>(Qb, Kb, Vb, Xbuf);

  // Output projection -> fp32.
  f32_to_bf16_kernel<<<nw4 / 256, 256, 0, stream>>>(Wo, Wbuf, nw4);
  gemm_bf16_kernel<1><<<gGemm, 256, 0, stream>>>(Xbuf, Wbuf, bo, out);
}
